// ResLFEBlock_61538291417254
// MI455X (gfx1250) — matrix operand order repaired and checked by reading, not compiled, not run on hardware
//
#include <hip/hip_runtime.h>
#include <hip/hip_bf16.h>

#define C_DIM   128
#define HID_DIM 512
#define NPTS    4096
#define BATCH   8
#define KNN     16
#define M_TOT   (BATCH * NPTS)   // 32768 rows
#define BN_EPS  1e-5f

typedef __attribute__((ext_vector_type(16))) __bf16 v16bf;
typedef __attribute__((ext_vector_type(8)))  float  v8f;

// ---------------------------------------------------------------------------
// WMMA wrapper: D = A(16x32 bf16) * B(32x16 bf16) + C(16x16 f32)
// ---------------------------------------------------------------------------
__device__ __forceinline__ v8f wmma_bf16(v16bf a, v16bf b, v8f c) {
  return __builtin_amdgcn_wmma_f32_16x16x32_bf16(
      /*neg_a=*/false, a, /*neg_b=*/false, b,
      /*c_mod=*/(short)0, c, /*reuse_a=*/false, /*reuse_b=*/false);
}

// A fragment (16x32) from f32 row-major tile, converted to bf16.
// ISA layout: lane<16 -> row=lane, K chunks {0..7, 16..23};
//             lane>=16 -> row=lane-16, K chunks {8..15, 24..31}.
__device__ __forceinline__ v16bf load_a_f32(const float* __restrict__ tile,
                                            int ld, int lane) {
  int r  = lane & 15;
  int hi = (lane >> 4) * 8;
  const float* p0 = tile + (size_t)r * ld + hi;
  const float* p1 = p0 + 16;
  v16bf a;
#pragma unroll
  for (int i = 0; i < 8; ++i) { a[i] = (__bf16)p0[i]; a[8 + i] = (__bf16)p1[i]; }
  return a;
}

// A fragment (16x32) from bf16 row-major LDS tile.
__device__ __forceinline__ v16bf load_a_lds(const __bf16* tile, int ld, int lane) {
  int r  = lane & 15;
  int hi = (lane >> 4) * 8;
  const __bf16* p0 = tile + (size_t)r * ld + hi;
  const __bf16* p1 = p0 + 16;
  v16bf a;
#pragma unroll
  for (int i = 0; i < 8; ++i) { a[i] = p0[i]; a[8 + i] = p1[i]; }
  return a;
}

// B fragment (32x16) from bf16 W^T (N x K row-major), wt points at (colTile*16, kbase).
// lane<16 -> col=lane, K {0..7, 16..23} ; lane>=16 -> col=lane-16, K {8..15, 24..31}.
__device__ __forceinline__ v16bf load_b(const __bf16* __restrict__ wt,
                                        int ldk, int lane) {
  int n  = lane & 15;
  int hi = (lane >> 4) * 8;
  typedef __attribute__((ext_vector_type(8))) __bf16 v8bf;
  const __bf16* p = wt + (size_t)n * ldk + hi;
  v8bf lo8 = *(const v8bf*)p;
  v8bf hi8 = *(const v8bf*)(p + 16);
  return __builtin_shufflevector(lo8, hi8, 0,1,2,3,4,5,6,7,8,9,10,11,12,13,14,15);
}

// ---------------------------------------------------------------------------
// Weight prep: W (K x N, f32 row-major) -> W^T (N x K, bf16 row-major)
// ---------------------------------------------------------------------------
__global__ void transpose_bf16(const float* __restrict__ W,
                               __bf16* __restrict__ WT, int K, int N) {
  int i = blockIdx.x * blockDim.x + threadIdx.x;
  if (i < K * N) {
    int n = i / K;
    int k = i - n * K;
    WT[i] = (__bf16)W[(size_t)k * N + n];
  }
}

// ---------------------------------------------------------------------------
// Fused FFN: xout = xin + BN(relu(xin*W1 + b1)*W2 + b2)
// 2 waves/block, 32 rows/wave (two 16-row tiles sharing every B fragment).
// In-place safe (each block touches only its own 64 rows).
// ---------------------------------------------------------------------------
__global__ __launch_bounds__(64)
void ffn_fused(const float* __restrict__ xin, float* __restrict__ xout,
               const __bf16* __restrict__ w1T, const float* __restrict__ b1,
               const __bf16* __restrict__ w2T, const float* __restrict__ b2,
               const float* __restrict__ gam, const float* __restrict__ bet,
               const float* __restrict__ mu,  const float* __restrict__ var) {
  __shared__ __bf16 Hs[2 * 32 * HID_DIM];   // 64 KB
  const int lane = threadIdx.x & 31;
  const int wave = threadIdx.x >> 5;
  const int rowbase = blockIdx.x * 64 + wave * 32;
  __bf16* Hw = Hs + wave * 32 * HID_DIM;

  const int nloc = lane & 15;
  const int mofs = (lane >> 4) * 8;

  // A fragments: two 16-row tiles, K=128 -> 4 fragments each
  v16bf A[2][4];
#pragma unroll
  for (int t = 0; t < 2; ++t)
#pragma unroll
    for (int kk = 0; kk < 4; ++kk)
      A[t][kk] = load_a_f32(xin + (size_t)(rowbase + t * 16) * C_DIM + kk * 32,
                            C_DIM, lane);

  // GEMM1: H = relu(x*W1 + b1) -> bf16 in LDS (32 x 512 per wave)
#pragma unroll 2
  for (int nt = 0; nt < HID_DIM / 16; ++nt) {
    v8f acc0 = {}, acc1 = {};
#pragma unroll
    for (int kk = 0; kk < 4; ++kk) {
      v16bf b = load_b(w1T + (size_t)(nt * 16) * C_DIM + kk * 32, C_DIM, lane);
      acc0 = wmma_bf16(A[0][kk], b, acc0);
      acc1 = wmma_bf16(A[1][kk], b, acc1);
    }
    float bias = b1[nt * 16 + nloc];
#pragma unroll
    for (int r = 0; r < 8; ++r) {
      float h0 = acc0[r] + bias;
      float h1 = acc1[r] + bias;
      h0 = h0 > 0.0f ? h0 : 0.0f;
      h1 = h1 > 0.0f ? h1 : 0.0f;
      Hw[(size_t)(mofs + r) * HID_DIM + nt * 16 + nloc]        = (__bf16)h0;
      Hw[(size_t)(16 + mofs + r) * HID_DIM + nt * 16 + nloc]   = (__bf16)h1;
    }
  }

  __syncthreads();   // per-wave LDS regions; conservative ordering

  // GEMM2: y = H*W2 ; 2 row tiles x 8 col tiles accumulated over K=512
  v8f acc2[2][8] = {};
  for (int kk = 0; kk < HID_DIM / 32; ++kk) {
    v16bf a0 = load_a_lds(Hw + kk * 32, HID_DIM, lane);
    v16bf a1 = load_a_lds(Hw + (size_t)16 * HID_DIM + kk * 32, HID_DIM, lane);
#pragma unroll
    for (int ct = 0; ct < 8; ++ct) {
      v16bf b = load_b(w2T + (size_t)(ct * 16) * HID_DIM + kk * 32, HID_DIM, lane);
      acc2[0][ct] = wmma_bf16(a0, b, acc2[0][ct]);
      acc2[1][ct] = wmma_bf16(a1, b, acc2[1][ct]);
    }
  }

  // Epilogue: bias + BatchNorm(eval) + residual
#pragma unroll
  for (int ct = 0; ct < 8; ++ct) {
    int col = ct * 16 + nloc;
    float sc = gam[col] * rsqrtf(var[col] + BN_EPS);
    float mb = mu[col];
    float bt = bet[col];
    float bb = b2[col];
#pragma unroll
    for (int t = 0; t < 2; ++t) {
#pragma unroll
      for (int r = 0; r < 8; ++r) {
        size_t idx = (size_t)(rowbase + t * 16 + mofs + r) * C_DIM + col;
        float y = acc2[t][ct][r] + bb;
        xout[idx] = xin[idx] + (y - mb) * sc + bt;
      }
    }
  }
}

// ---------------------------------------------------------------------------
// VFR linear: h = x * Wv + b   (C x C GEMM), f32 output to workspace.
// 8 waves/block, 32 rows/wave (two tiles sharing each B fragment).
// ---------------------------------------------------------------------------
__global__ __launch_bounds__(256)
void vfr_linear(const float* __restrict__ x, const __bf16* __restrict__ wT,
                const float* __restrict__ b, float* __restrict__ h) {
  const int lane = threadIdx.x & 31;
  const int wave = threadIdx.x >> 5;
  const int rowbase = blockIdx.x * 256 + wave * 32;
  const int nloc = lane & 15;
  const int mofs = (lane >> 4) * 8;

  v16bf A[2][4];
#pragma unroll
  for (int t = 0; t < 2; ++t)
#pragma unroll
    for (int kk = 0; kk < 4; ++kk)
      A[t][kk] = load_a_f32(x + (size_t)(rowbase + t * 16) * C_DIM + kk * 32,
                            C_DIM, lane);

#pragma unroll
  for (int ct = 0; ct < 8; ++ct) {
    v8f acc0 = {}, acc1 = {};
#pragma unroll
    for (int kk = 0; kk < 4; ++kk) {
      v16bf b2 = load_b(wT + (size_t)(ct * 16) * C_DIM + kk * 32, C_DIM, lane);
      acc0 = wmma_bf16(A[0][kk], b2, acc0);
      acc1 = wmma_bf16(A[1][kk], b2, acc1);
    }
    int col = ct * 16 + nloc;
    float bb = b[col];
#pragma unroll
    for (int r = 0; r < 8; ++r) {
      h[(size_t)(rowbase + mofs + r) * C_DIM + col]      = acc0[r] + bb;
      h[(size_t)(rowbase + 16 + mofs + r) * C_DIM + col] = acc1[r] + bb;
    }
  }
}

// ---------------------------------------------------------------------------
// VFR gather: x += BN( max_k( h[knn[k]] - h[self] ) )
// 2 rows per 256-thread block; one lane per channel; idx staged in LDS.
// ---------------------------------------------------------------------------
__global__ __launch_bounds__(256)
void vfr_gather_bn(float* __restrict__ x, const float* __restrict__ h,
                   const long long* __restrict__ knn,
                   const float* __restrict__ gam, const float* __restrict__ bet,
                   const float* __restrict__ mu,  const float* __restrict__ var) {
  __shared__ int sidx[2][KNN];
  const int sub = threadIdx.x >> 7;       // 0..1 : row within block
  const int c   = threadIdx.x & 127;      // channel
  const int row = blockIdx.x * 2 + sub;   // global point row in [0, B*N)
  const int b   = row >> 12;              // row / 4096

  if (c < KNN) sidx[sub][c] = (int)knn[(size_t)row * KNN + c];
  __syncthreads();

  const float* hb = h + (size_t)b * NPTS * C_DIM;
  const float hs = h[(size_t)row * C_DIM + c];
  float acc = -3.4e38f;
#pragma unroll
  for (int k = 0; k < KNN; ++k) {
    float nb = hb[(size_t)sidx[sub][k] * C_DIM + c];
    acc = fmaxf(acc, nb - hs);
  }
  float bn = (acc - mu[c]) * (gam[c] * rsqrtf(var[c] + BN_EPS)) + bet[c];
  x[(size_t)row * C_DIM + c] += bn;
}

// ---------------------------------------------------------------------------
// Orchestration
// ---------------------------------------------------------------------------
extern "C" void kernel_launch(void* const* d_in, const int* in_sizes, int n_in,
                              void* d_out, int out_size, void* d_ws, size_t ws_size,
                              hipStream_t stream) {
  const float*     x         = (const float*)d_in[0];
  const long long* knn       = (const long long*)d_in[1];
  const float*     mlp0_w1   = (const float*)d_in[2];
  const float*     mlp0_b1   = (const float*)d_in[3];
  const float*     mlp0_w2   = (const float*)d_in[4];
  const float*     mlp0_b2   = (const float*)d_in[5];
  const float*     mlp0_g    = (const float*)d_in[6];
  const float*     mlp0_beta = (const float*)d_in[7];
  const float*     mlp0_m    = (const float*)d_in[8];
  const float*     mlp0_v    = (const float*)d_in[9];
  const float*     vfr_w     = (const float*)d_in[10];
  const float*     vfr_b     = (const float*)d_in[11];
  const float*     vfr_g     = (const float*)d_in[12];
  const float*     vfr_beta  = (const float*)d_in[13];
  const float*     vfr_m     = (const float*)d_in[14];
  const float*     vfr_v     = (const float*)d_in[15];
  const float*     ffn_w1    = (const float*)d_in[16];
  const float*     ffn_b1    = (const float*)d_in[17];
  const float*     ffn_w2    = (const float*)d_in[18];
  const float*     ffn_b2    = (const float*)d_in[19];
  const float*     ffn_g     = (const float*)d_in[20];
  const float*     ffn_beta  = (const float*)d_in[21];
  const float*     ffn_m     = (const float*)d_in[22];
  const float*     ffn_v     = (const float*)d_in[23];

  float* out = (float*)d_out;
  char*  ws  = (char*)d_ws;

  // Workspace layout
  float*  h     = (float*)ws;                        // 32768*128*4 = 16,777,216 B
  __bf16* w1T0  = (__bf16*)(ws + 16777216);          //   131,072 B
  __bf16* w2T0  = (__bf16*)(ws + 16908288);          //   131,072 B
  __bf16* vwT   = (__bf16*)(ws + 17039360);          //    65,536 B (2 layers)
  __bf16* fw1T  = (__bf16*)(ws + 17104896);          //   262,144 B (2 layers)
  __bf16* fw2T  = (__bf16*)(ws + 17367040);          //   262,144 B (2 layers)

  const dim3 tb(256);
  auto gblk = [](int n) { return dim3((n + 255) / 256); };

  // Weight prep (bf16, transposed to N x K)
  transpose_bf16<<<gblk(C_DIM * HID_DIM), tb, 0, stream>>>(mlp0_w1, w1T0, C_DIM, HID_DIM);
  transpose_bf16<<<gblk(HID_DIM * C_DIM), tb, 0, stream>>>(mlp0_w2, w2T0, HID_DIM, C_DIM);
  for (int i = 0; i < 2; ++i) {
    transpose_bf16<<<gblk(C_DIM * C_DIM), tb, 0, stream>>>(
        vfr_w + (size_t)i * C_DIM * C_DIM, vwT + (size_t)i * C_DIM * C_DIM, C_DIM, C_DIM);
    transpose_bf16<<<gblk(C_DIM * HID_DIM), tb, 0, stream>>>(
        ffn_w1 + (size_t)i * C_DIM * HID_DIM, fw1T + (size_t)i * C_DIM * HID_DIM, C_DIM, HID_DIM);
    transpose_bf16<<<gblk(HID_DIM * C_DIM), tb, 0, stream>>>(
        ffn_w2 + (size_t)i * HID_DIM * C_DIM, fw2T + (size_t)i * HID_DIM * C_DIM, HID_DIM, C_DIM);
  }

  // Stage 0: x = x + ffn_mlp0(x)   (reads d_in, writes d_out)
  ffn_fused<<<M_TOT / 64, 64, 0, stream>>>(x, out, w1T0, mlp0_b1, w2T0, mlp0_b2,
                                           mlp0_g, mlp0_beta, mlp0_m, mlp0_v);

  for (int i = 0; i < 2; ++i) {
    vfr_linear<<<M_TOT / 256, 256, 0, stream>>>(out, vwT + (size_t)i * C_DIM * C_DIM,
                                                vfr_b + (size_t)i * C_DIM, h);
    vfr_gather_bn<<<M_TOT / 2, 256, 0, stream>>>(out, h, knn,
                                                 vfr_g + (size_t)i * C_DIM,
                                                 vfr_beta + (size_t)i * C_DIM,
                                                 vfr_m + (size_t)i * C_DIM,
                                                 vfr_v + (size_t)i * C_DIM);
    ffn_fused<<<M_TOT / 64, 64, 0, stream>>>(out, out,
                                             fw1T + (size_t)i * C_DIM * HID_DIM,
                                             ffn_b1 + (size_t)i * HID_DIM,
                                             fw2T + (size_t)i * HID_DIM * C_DIM,
                                             ffn_b2 + (size_t)i * C_DIM,
                                             ffn_g + (size_t)i * C_DIM,
                                             ffn_beta + (size_t)i * C_DIM,
                                             ffn_m + (size_t)i * C_DIM,
                                             ffn_v + (size_t)i * C_DIM);
  }
}
